// AdaptivePoolCompressor_19284403159434
// MI455X (gfx1250) — matrix operand order repaired and checked by reading, not compiled, not run on hardware
//
#include <hip/hip_runtime.h>
#include <cstdint>
#include <cstddef>

// ---------------------------------------------------------------------------
// AdaptivePoolCompressor for MI455X (gfx1250, wave32, WMMA).
//   x:[4,8192,1024] f32, W1:[1024,256], b1:[256], W2:[256,1], b2:[1], pp:[512]
//   out: compressed [4,512,1024] f32  ++  mask [4,512] f32 (ones)
//
// Pipeline:
//   0) pack:    x -> bf16 hi/lo B-fragments (K=S,N=D); W1 -> bf16 B-fragments
//   1) logits:  GELU(x@W1+b1)@W2+b2 via WMMA (x split hi/lo, W1 hi-only)
//   2) softmax: w10 = 10*softmax(logits) per batch (+ mask = 1)
//   3) epack:   banded-softmax numerators e = exp(combined-10) computed ONCE,
//               packed as A-fragments (hi/lo) + per-row partial denominators
//   4) pool:    pure streaming WMMA GEMM (3 split-WMMAs/tile) + normalize
//
// Workspace (~195.5 MB):
//   [0,128MB)          xp  : x packed (hi 1KB + lo 1KB per 32x16 tile)
//   [128MB,129MB)      w1p : W1 packed
//   [129MB,193MB)      ea  : A-fragment blobs for pooling GEMM (hi/lo)
//   [193MB,195MB)      dn  : denominator partials [4][32][16][256] f32
//   [195MB,+128KB)     logits f32
//   [+128KB,+256KB)    w10 f32
// ---------------------------------------------------------------------------

#define BB 4
#define SS 8192
#define DD 1024
#define HH 256
#define TT 512

#define KT2 (SS / 32)   // 256 K-tiles for pooling GEMM
#define DT2 (DD / 16)   // 64 N-tiles (d)
#define KT1 (DD / 32)   // 32 K-tiles for predictor GEMM
#define NT1 (HH / 16)   // 16 N-tiles (hidden)

typedef __attribute__((ext_vector_type(16))) __bf16 v16bf;
typedef __attribute__((ext_vector_type(8)))  float  v8f;

union Frag {
    v16bf v;
    unsigned short u[16];
    uint4 q[2];
};

static __device__ __forceinline__ unsigned short f2bf(float f) {
    unsigned int u = __float_as_uint(f);
    unsigned int r = u + 0x7FFFu + ((u >> 16) & 1u);   // round-to-nearest-even
    return (unsigned short)(r >> 16);
}
static __device__ __forceinline__ float bf2f(unsigned short h) {
    return __uint_as_float(((unsigned int)h) << 16);
}
static __device__ __forceinline__ void put2(Frag& hi, Frag& lo, int i, float f) {
    unsigned short h = f2bf(f);
    hi.u[i] = h;
    lo.u[i] = f2bf(f - bf2f(h));
}
static __device__ __forceinline__ uint4 pack8(const unsigned short* h) {
    uint4 q;
    q.x = (unsigned)h[0] | ((unsigned)h[1] << 16);
    q.y = (unsigned)h[2] | ((unsigned)h[3] << 16);
    q.z = (unsigned)h[4] | ((unsigned)h[5] << 16);
    q.w = (unsigned)h[6] | ((unsigned)h[7] << 16);
    return q;
}

#define WMMA_BF16(C, A, B) \
    (C) = __builtin_amdgcn_wmma_f32_16x16x32_bf16(false, (A), false, (B), (short)0, (C), false, false)

// ---------------------------------------------------------------------------
// Kernel 0: pack x and W1 into bf16 hi/lo B-fragment blobs.
// B-frag (32x16, KxN): lane L holds column n = L%16, K = (L/16)*16 + j.
// Blob = [hi: 1KB][lo: 1KB].
// ---------------------------------------------------------------------------
__global__ __launch_bounds__(256) void apc_pack_kernel(
    const float* __restrict__ x, const float* __restrict__ W1,
    unsigned char* __restrict__ xp, unsigned char* __restrict__ w1p)
{
    const int tid = blockIdx.x * 256 + threadIdx.x;
    const int NXP = BB * KT2 * DT2 * 64;      // 4,194,304 threads for x
    const int NWP = KT1 * NT1 * 64;           // 32,768 threads for W1
    unsigned short hs[8], ls[8];
    if (tid < NXP) {
        int idx  = tid;
        int j8   = idx & 1;
        int lane = (idx >> 1) & 31;
        int dt   = (idx >> 6) & 63;
        int kt   = (idx >> 12) & 255;
        int b    = idx >> 20;
        int d    = dt * 16 + (lane & 15);
        int sbase = kt * 32 + (lane >> 4) * 16 + j8 * 8;
#pragma unroll
        for (int j = 0; j < 8; ++j) {
            float f = x[((size_t)b * SS + (sbase + j)) * DD + d];
            hs[j] = f2bf(f);
            ls[j] = f2bf(f - bf2f(hs[j]));
        }
        unsigned char* blob = xp + ((size_t)((b * KT2 + kt) * DT2 + dt)) * 2048;
        *(uint4*)(blob + lane * 32 + j8 * 16)        = pack8(hs);
        *(uint4*)(blob + 1024 + lane * 32 + j8 * 16) = pack8(ls);
    } else if (tid < NXP + NWP) {
        int idx  = tid - NXP;
        int j8   = idx & 1;
        int lane = (idx >> 1) & 31;
        int nt   = (idx >> 6) & 15;
        int kt   = idx >> 10;
        int n    = nt * 16 + (lane & 15);
        int kbase = kt * 32 + (lane >> 4) * 16 + j8 * 8;
#pragma unroll
        for (int j = 0; j < 8; ++j) {
            float f = W1[(size_t)(kbase + j) * HH + n];
            hs[j] = f2bf(f);
        }
        unsigned char* blob = w1p + (size_t)(kt * NT1 + nt) * 1024;  // hi only
        *(uint4*)(blob + lane * 32 + j8 * 16) = pack8(hs);
    }
}

// ---------------------------------------------------------------------------
// Kernel 1: logits = GELU(x@W1 + b1)@W2 + b2 via WMMA.
// Block = 8 waves; each wave owns 16 rows, full H=256 (16 accumulators).
// x is split hi/lo (2 WMMAs); W1 hi-only (its rounding contributes ~3e-3 to
// logits, which vanishes through softmax -> x10 -> softmax).
// A-fragment (16x32 bf16) in this code's contiguous order: lane holds row m = lane%16,
// K offsets = half==0 ? {0..15} : {16..31}.
// ---------------------------------------------------------------------------
__global__ __launch_bounds__(256) void apc_logits_kernel(
    const float* __restrict__ x, const unsigned char* __restrict__ w1p,
    const float* __restrict__ b1, const float* __restrict__ W2,
    const float* __restrict__ b2, float* __restrict__ logits)
{
    const int tid = threadIdx.x, wave = tid >> 5, lane = tid & 31;
    const int half = lane >> 4, half8 = half * 8, nloc = lane & 15;
    const int R0 = blockIdx.x * 128 + wave * 16;             // global row base
    const float* xrow = x + (size_t)(R0 + nloc) * DD;

    v8f c[16];
#pragma unroll
    for (int nt = 0; nt < 16; ++nt) { v8f z = {}; c[nt] = z; }

    for (int kt = 0; kt < KT1; ++kt) {
        const int kbase = kt * 32;
        const float4* p1 = (const float4*)(xrow + kbase + half * 16);
        const float4* p2 = (const float4*)(xrow + kbase + half * 16 + 8);
        float4 f0 = p1[0], f1 = p1[1], f2 = p2[0], f3 = p2[1];
        Frag ahi, alo;
        put2(ahi, alo, 0, f0.x);  put2(ahi, alo, 1, f0.y);
        put2(ahi, alo, 2, f0.z);  put2(ahi, alo, 3, f0.w);
        put2(ahi, alo, 4, f1.x);  put2(ahi, alo, 5, f1.y);
        put2(ahi, alo, 6, f1.z);  put2(ahi, alo, 7, f1.w);
        put2(ahi, alo, 8, f2.x);  put2(ahi, alo, 9, f2.y);
        put2(ahi, alo, 10, f2.z); put2(ahi, alo, 11, f2.w);
        put2(ahi, alo, 12, f3.x); put2(ahi, alo, 13, f3.y);
        put2(ahi, alo, 14, f3.z); put2(ahi, alo, 15, f3.w);
#pragma unroll
        for (int nt = 0; nt < NT1; ++nt) {
            const uint4* bph = (const uint4*)(w1p + (size_t)(kt * NT1 + nt) * 1024 + lane * 32);
            Frag bhi;
            bhi.q[0] = bph[0]; bhi.q[1] = bph[1];
            WMMA_BF16(c[nt], ahi.v, bhi.v);
            WMMA_BF16(c[nt], alo.v, bhi.v);
        }
    }

    float acc[8];
#pragma unroll
    for (int r = 0; r < 8; ++r) acc[r] = 0.0f;
    const float b2v = b2[0];
#pragma unroll
    for (int nt = 0; nt < NT1; ++nt) {
        const int n = nt * 16 + nloc;
        const float b1v = b1[n], w2v = W2[n];
#pragma unroll
        for (int r = 0; r < 8; ++r) {
            float h = c[nt][r] + b1v;
            float g = 0.5f * h * (1.0f + erff(h * 0.7071067811865476f));
            acc[r] += g * w2v;
        }
    }
#pragma unroll
    for (int r = 0; r < 8; ++r) {
        acc[r] += __shfl_xor(acc[r], 1);
        acc[r] += __shfl_xor(acc[r], 2);
        acc[r] += __shfl_xor(acc[r], 4);
        acc[r] += __shfl_xor(acc[r], 8);
    }
    if (nloc == 0) {
#pragma unroll
        for (int r = 0; r < 8; ++r)
            logits[R0 + half * 8 + r] = acc[r] + b2v;   // row M = r + 8*half
    }
}

// ---------------------------------------------------------------------------
// Kernel 2: w10 = 10 * softmax(logits, axis=S) per batch; also writes mask=1.
// ---------------------------------------------------------------------------
__global__ __launch_bounds__(256) void apc_softmax_kernel(
    const float* __restrict__ logits, float* __restrict__ w10,
    float* __restrict__ mask)
{
    __shared__ float red[256];
    const int b = blockIdx.x, tid = threadIdx.x;
    const float* lg = logits + (size_t)b * SS;

    float lmax = -3.4e38f;
    for (int i = tid; i < SS; i += 256) lmax = fmaxf(lmax, lg[i]);
    red[tid] = lmax; __syncthreads();
    for (int s = 128; s > 0; s >>= 1) {
        if (tid < s) red[tid] = fmaxf(red[tid], red[tid + s]);
        __syncthreads();
    }
    const float bmax = red[0]; __syncthreads();

    float lsum = 0.0f;
    for (int i = tid; i < SS; i += 256) lsum += expf(lg[i] - bmax);
    red[tid] = lsum; __syncthreads();
    for (int s = 128; s > 0; s >>= 1) {
        if (tid < s) red[tid] += red[tid + s];
        __syncthreads();
    }
    const float inv = 10.0f / red[0];

    float* wp = w10 + (size_t)b * SS;
    for (int i = tid; i < SS; i += 256) wp[i] = expf(lg[i] - bmax) * inv;

    for (int i = tid; i < TT; i += 256) mask[b * TT + i] = 1.0f;
}

// ---------------------------------------------------------------------------
// Kernel 3: epack -- compute e = exp(combined - 10) ONCE per (b,t,s), packed
// straight into 16x32 A-fragment hi/lo blobs, plus per-row partial denominator
// sums dn[b][tt][m][kt] (deterministic fixed-order reduction later).
// combined = -|pp_t - s/(S-1)|*S + w10[b,s] <= ~10, so a fixed shift of 10 is
// overflow-free and preserves the exact softmax after normalization.
// One wave per blob; 8 blobs per block.
// ---------------------------------------------------------------------------
__global__ __launch_bounds__(256) void apc_epack_kernel(
    const float* __restrict__ w10, const float* __restrict__ pp,
    unsigned char* __restrict__ ea, float* __restrict__ dn)
{
    const int wave = threadIdx.x >> 5, lane = threadIdx.x & 31;
    const int id = blockIdx.x * 8 + wave;        // ((b*32+tt)*256 + kt)
    const int kt = id & 255;
    const int tt = (id >> 8) & 31;
    const int b  = id >> 13;
    const int half = lane >> 4, half8 = half * 8, nloc = lane & 15;

    const float ppv = pp[tt * 16 + nloc];
    const float INVP = 1.0f / (float)(SS - 1);
    const float* wrow = w10 + (size_t)b * SS;
    const int kbase = kt * 32;

    Frag ahi, alo;
    float dsum = 0.0f;
#pragma unroll
    for (int e = 0; e < 16; ++e) {
        const int koff = half * 16 + e;
        const int s = kbase + koff;
        const float comb = wrow[s] - fabsf(ppv - (float)s * INVP) * (float)SS;
        const float ex = __expf(comb - 10.0f);
        dsum += ex;
        put2(ahi, alo, e, ex);
    }
    unsigned char* blob = ea + (size_t)id * 2048;
    *(uint4*)(blob + lane * 32)        = ahi.q[0];
    *(uint4*)(blob + lane * 32 + 16)   = ahi.q[1];
    *(uint4*)(blob + 1024 + lane * 32)      = alo.q[0];
    *(uint4*)(blob + 1024 + lane * 32 + 16) = alo.q[1];

    const float ds2 = dsum + __shfl_xor(dsum, 16);   // denom[m] partial for kt
    if (lane < 16)
        dn[((size_t)(b * 32 + tt) * 16 + lane) * 256 + kt] = ds2;
}

// ---------------------------------------------------------------------------
// Kernel 4: pool -- pure streaming WMMA GEMM with split precision:
// C += Ahi*Bhi + Ahi*Blo + Alo*Bhi over K=S, then normalize by denominators.
// Block = (b, 16-row t-tile); 8 waves x 8 d-tiles = full D=1024.
// ---------------------------------------------------------------------------
__global__ __launch_bounds__(256) void apc_pool_kernel(
    const unsigned char* __restrict__ xp, const unsigned char* __restrict__ ea,
    const float* __restrict__ dn, float* __restrict__ out)
{
    __shared__ float dred[256];
    __shared__ float dinv[16];
    const int b = blockIdx.x >> 5;
    const int tt = blockIdx.x & 31;
    const int tid = threadIdx.x, wave = tid >> 5, lane = tid & 31;
    const int half = lane >> 4, nloc = lane & 15;

    // Deterministic denominator reduction: denom[m] = sum_kt dn[b][tt][m][kt]
    {
        const int m = tid >> 4, j = tid & 15;
        const float* dp = dn + ((size_t)(b * 32 + tt) * 16 + m) * 256 + j * 16;
        float s = 0.0f;
#pragma unroll
        for (int jj = 0; jj < 16; ++jj) s += dp[jj];
        dred[tid] = s;
    }
    __syncthreads();
    if (tid < 16) {
        float t = 0.0f;
#pragma unroll
        for (int jj = 0; jj < 16; ++jj) t += dred[tid * 16 + jj];
        dinv[tid] = 1.0f / t;
    }
    __syncthreads();

    v8f c[8];
#pragma unroll
    for (int t = 0; t < 8; ++t) { v8f z = {}; c[t] = z; }

    const unsigned char* ablob0 = ea + (size_t)((b * 32 + tt) * 256) * 2048;
    const size_t bwave = (size_t)(b * KT2) * DT2 + wave * 8;

    for (int kt = 0; kt < KT2; ++kt) {
        const unsigned char* ablob = ablob0 + (size_t)kt * 2048;
        Frag ahi, alo;
        {
            const uint4* aph = (const uint4*)(ablob + lane * 32);
            const uint4* apl = (const uint4*)(ablob + 1024 + lane * 32);
            ahi.q[0] = aph[0]; ahi.q[1] = aph[1];
            alo.q[0] = apl[0]; alo.q[1] = apl[1];
        }
        if (kt + 1 < KT2) {
            __builtin_prefetch(ablob + 2048 + lane * 64, 0, 1);  // next A blob
        }
        const unsigned char* brow = xp + (bwave + (size_t)kt * DT2) * 2048;
#pragma unroll
        for (int tl = 0; tl < 8; ++tl) {
            const unsigned char* blob = brow + (size_t)tl * 2048;
            const uint4* bph = (const uint4*)(blob + lane * 32);
            const uint4* bpl = (const uint4*)(blob + 1024 + lane * 32);
            Frag bhi, blo;
            bhi.q[0] = bph[0]; bhi.q[1] = bph[1];
            blo.q[0] = bpl[0]; blo.q[1] = bpl[1];
            if (kt + 1 < KT2) {
                __builtin_prefetch(blob + (size_t)DT2 * 2048 + lane * 64, 0, 1);
            }
            WMMA_BF16(c[tl], ahi.v, bhi.v);
            WMMA_BF16(c[tl], ahi.v, blo.v);
            WMMA_BF16(c[tl], alo.v, bhi.v);
        }
    }

    float invd[8];
#pragma unroll
    for (int r = 0; r < 8; ++r) invd[r] = dinv[r + 8 * half];  // row M = r+8*half

    const int dbase = wave * 128 + nloc;
#pragma unroll
    for (int tl = 0; tl < 8; ++tl) {
#pragma unroll
        for (int r = 0; r < 8; ++r) {
            const int trow = tt * 16 + r + 8 * half;
            out[((size_t)(b * TT + trow)) * DD + dbase + tl * 16] = c[tl][r] * invd[r];
        }
    }
}

// ---------------------------------------------------------------------------
extern "C" void kernel_launch(void* const* d_in, const int* in_sizes, int n_in,
                              void* d_out, int out_size, void* d_ws, size_t ws_size,
                              hipStream_t stream) {
    (void)in_sizes; (void)n_in; (void)out_size; (void)ws_size;
    const float* x  = (const float*)d_in[0];
    const float* W1 = (const float*)d_in[1];
    const float* b1 = (const float*)d_in[2];
    const float* W2 = (const float*)d_in[3];
    const float* b2 = (const float*)d_in[4];
    const float* pp = (const float*)d_in[5];

    unsigned char* ws = (unsigned char*)d_ws;
    unsigned char* xp     = ws;                                  // 128 MB
    unsigned char* w1p    = ws + ((size_t)128u << 20);           // 512 KB used
    unsigned char* ea     = ws + ((size_t)129u << 20);           // 64 MB
    float*         dn     = (float*)(ws + ((size_t)193u << 20)); // 2 MB
    float*         logits = (float*)(ws + ((size_t)195u << 20)); // 128 KB
    float*         w10    = logits + BB * SS;                    // 128 KB

    float* outc = (float*)d_out;
    float* mask = outc + (size_t)BB * TT * DD;

    const int NXP = BB * KT2 * DT2 * 64;
    const int NWP = KT1 * NT1 * 64;
    const int packBlocks = (NXP + NWP + 255) / 256;

    apc_pack_kernel<<<packBlocks, 256, 0, stream>>>(x, W1, xp, w1p);
    apc_logits_kernel<<<(BB * SS) / 128, 256, 0, stream>>>(x, w1p, b1, W2, b2, logits);
    apc_softmax_kernel<<<BB, 256, 0, stream>>>(logits, w10, mask);
    apc_epack_kernel<<<(BB * 32 * KT2) / 8, 256, 0, stream>>>(w10, pp, ea, dn);
    apc_pool_kernel<<<BB * (TT / 16), 256, 0, stream>>>(xp, ea, dn, outc);
}
